// MyModel_33930241638744
// MI455X (gfx1250) — matrix operand order repaired and checked by reading, not compiled, not run on hardware
//
#include <hip/hip_runtime.h>

typedef __bf16 bf16;
typedef __attribute__((ext_vector_type(16))) __bf16 v16bf;
typedef __attribute__((ext_vector_type(8)))  __bf16 v8bf;
typedef __attribute__((ext_vector_type(8)))  float  v8f;
typedef __attribute__((ext_vector_type(4)))  float  v4f;

// ---------- helpers ----------
__device__ __forceinline__ bf16 cvt_bf16(float f) {
  unsigned u = __float_as_uint(f);
  u += 0x7fffu + ((u >> 16) & 1u);                 // round-to-nearest-even
  unsigned short h = (unsigned short)(u >> 16);
  return __builtin_bit_cast(bf16, h);
}
__device__ __forceinline__ unsigned f2ord(float f) {  // order-preserving uint
  unsigned u = __float_as_uint(f);
  return (u >> 31) ? ~u : (u | 0x80000000u);
}
__device__ __forceinline__ float ord2f(unsigned u) {
  return (u >> 31) ? __uint_as_float(u & 0x7fffffffu) : __uint_as_float(~u);
}

// ---------- generic WMMA GEMM: C[M,Nout] = A[M,K] * Bpacked + bias ----------
// Bpacked layout (per 16-col tile t, per 32-k step s, per lane, 16 bf16):
//   col = lane&15 ; k = s*32 + (lane>>4)*8 + (i&7) + (i>>3)*16   (one contiguous 32B load)
// NT is padded to a multiple of NTW with zero tiles -> branch-free K loop.
// Rows are clamped to M-1 (always in-bounds); out-of-range rows only pollute
// D rows >= M, which the store guard drops.  No divergence in the K loop.
// A fragment (16-bit A 16x32 ISA layout): row = lane&15, khalf = lane>>4,
//   elems 0..7  = A[row][k0 + khalf*8 + 0..7]
//   elems 8..15 = A[row][k0 + 16 + khalf*8 + 0..7]   (two contiguous 16B runs)
template <typename AT, typename OT, int NTW>
__global__ __launch_bounds__(128) void gemm_wmma(
    const AT* __restrict__ A, int lda,
    const bf16* __restrict__ Bp, int ksteps,
    const float* __restrict__ bias,
    OT* __restrict__ Cout, int ldc, int M, int Nout)
{
  const int lane  = threadIdx.x & 31;
  const int wave  = threadIdx.x >> 5;
  const int rlow  = lane & 15;
  const int khalf = lane >> 4;
  const int rowBase = (blockIdx.x * 4 + wave) * 16;
  const int r  = rowBase + rlow;
  const int rc = r < M ? r : M - 1;                // clamp: always in-bounds
  const int tg = blockIdx.y * NTW;

  v8f acc[NTW];
#pragma unroll
  for (int t = 0; t < NTW; ++t)
#pragma unroll
    for (int i = 0; i < 8; ++i) acc[t][i] = 0.f;

  const AT*   arow  = A + (size_t)rc * lda + khalf * 8;
  const bf16* bbase = Bp + (((size_t)tg * ksteps) * 32 + lane) * 16;
  const size_t btile = (size_t)ksteps * 512;       // elements per 16-col tile

  auto loadA = [&](int s, v16bf& af) {
    const AT* p0 = arow + s * 32;
    const AT* p1 = p0 + 16;
    __builtin_prefetch(p0 + 64, 0, 3);             // near-cache prefetch, 2 steps ahead
    if constexpr (sizeof(AT) == 4) {               // fp32 A, convert on the fly
      v4f f0 = *(const v4f*)(p0);
      v4f f1 = *(const v4f*)(p0 + 4);
      v4f f2 = *(const v4f*)(p1);
      v4f f3 = *(const v4f*)(p1 + 4);
#pragma unroll
      for (int i = 0; i < 4; ++i) {
        af[i]      = cvt_bf16(f0[i]);
        af[i + 4]  = cvt_bf16(f1[i]);
        af[i + 8]  = cvt_bf16(f2[i]);
        af[i + 12] = cvt_bf16(f3[i]);
      }
    } else {                                       // bf16 A, straight load
      v8bf b0 = *(const v8bf*)(p0);
      v8bf b1 = *(const v8bf*)(p1);
#pragma unroll
      for (int i = 0; i < 8; ++i) { af[i] = b0[i]; af[i + 8] = b1[i]; }
    }
  };
  auto loadB = [&](int s, v16bf* bset) {
#pragma unroll
    for (int tt = 0; tt < NTW; ++tt)
      bset[tt] = *(const v16bf*)(bbase + tt * btile + (size_t)s * 512);
  };
  auto mma = [&](const v16bf& af, const v16bf* bset) {
#pragma unroll
    for (int tt = 0; tt < NTW; ++tt)
      acc[tt] = __builtin_amdgcn_wmma_f32_16x16x32_bf16(
          false, af, false, bset[tt], (short)0, acc[tt], false, false);
  };

  // double-buffered, branch-free steady state; 1- or 2-step tail peeled
  v16bf afA, afB;
  v16bf bA[NTW], bB[NTW];
  loadA(0, afA);
  loadB(0, bA);

  int s = 0;
  for (; s + 2 < ksteps; s += 2) {
    loadA(s + 1, afB);
    loadB(s + 1, bB);
    mma(afA, bA);
    loadA(s + 2, afA);
    loadB(s + 2, bA);
    mma(afB, bB);
  }
  if (s + 1 < ksteps) {       // 2 steps remain
    loadA(s + 1, afB);
    loadB(s + 1, bB);
    mma(afA, bA);
    mma(afB, bB);
  } else {                    // 1 step remains
    mma(afA, bA);
  }

#pragma unroll
  for (int tt = 0; tt < NTW; ++tt) {
    int col = (tg + tt) * 16 + rlow;
    if (col >= Nout) continue;
    float bv = bias ? bias[col] : 0.f;
#pragma unroll
    for (int v = 0; v < 8; ++v) {                  // C/D layout: row = khalf*8+v
      int rr = rowBase + khalf * 8 + v;
      if (rr < M) {
        float val = acc[tt][v] + bv;
        if constexpr (sizeof(OT) == 2) Cout[(size_t)rr * ldc + col] = cvt_bf16(val);
        else                           Cout[(size_t)rr * ldc + col] = val;
      }
    }
  }
}

// ---------- pack fp32 weight [Kreal,Nreal] (ld=ldb) into B-fragment layout ----------
// NT may exceed ceil(Nreal/16): pad tiles are written as zeros.
__global__ void pack_b_kernel(const float* __restrict__ B, bf16* __restrict__ out,
                              int Kreal, int Nreal, int ldb, int ksteps, int NT)
{
  int idx = blockIdx.x * blockDim.x + threadIdx.x;
  int total = NT * ksteps * 512;
  if (idx >= total) return;
  int i    = idx & 15;
  int lane = (idx >> 4) & 31;
  int s    = (idx >> 9) % ksteps;
  int t    = (idx >> 9) / ksteps;
  int col  = t * 16 + (lane & 15);
  int k    = s * 32 + (lane >> 4) * 8 + (i & 7) + (i >> 3) * 16;
  float v  = (k < Kreal && col < Nreal) ? B[(size_t)k * ldb + col] : 0.f;
  out[idx] = cvt_bf16(v);
}

__global__ void padbias_kernel(const float* __restrict__ b, float* __restrict__ out,
                               int nreal, int npad)
{
  int i = blockIdx.x * blockDim.x + threadIdx.x;
  if (i < npad) out[i] = (i < nreal) ? b[i] : 0.f;
}

// build x350 (ld 352): cols 0..23 = feats[:, :24] (bf16), cols 350,351 = 0
__global__ void build24_kernel(const float* __restrict__ feats, int lda,
                               bf16* __restrict__ x350, int M)
{
  int i = blockIdx.x * blockDim.x + threadIdx.x;
  int n = i >> 5, j = i & 31;
  if (n >= M) return;
  if (j < 24)       x350[(size_t)n * 352 + j]        = cvt_bf16(feats[(size_t)n * lda + j]);
  else if (j < 26)  x350[(size_t)n * 352 + 326 + j]  = cvt_bf16(0.f);  // cols 350,351
}

// ---------- GAT per-node prep: s,d logit terms; init m,z,acc ----------
__global__ void sd_init_kernel(const float* __restrict__ h,
                               const float* __restrict__ asrc, const float* __restrict__ adst,
                               float* __restrict__ s, float* __restrict__ d,
                               unsigned* __restrict__ m, float* __restrict__ z,
                               float* __restrict__ acc, int N, int H)
{
  int n = blockIdx.x * blockDim.x + threadIdx.x;
  if (n >= N) return;
  int C = 32 / H;
  const float* hr = h + (size_t)n * 32;
  for (int hh = 0; hh < H; ++hh) {
    float ss = 0.f, dd = 0.f;
    for (int c = 0; c < C; ++c) {
      float hv = hr[hh * C + c];
      ss += hv * asrc[hh * C + c];
      dd += hv * adst[hh * C + c];
    }
    s[(size_t)n * H + hh] = ss;
    d[(size_t)n * H + hh] = dd;
    m[(size_t)n * H + hh] = 0x007FFFFFu;   // f2ord(-inf)
    z[(size_t)n * H + hh] = 0.f;
  }
#pragma unroll
  for (int j = 0; j < 32; ++j) acc[(size_t)n * 32 + j] = 0.f;
}

__device__ __forceinline__ void edge_sd(const int* __restrict__ ei, int E, long e,
                                        int& src, int& dst)
{
  if (e < E) { src = ei[e]; dst = ei[(size_t)E + e]; }
  else       { src = (int)(e - E); dst = src; }      // self loops
}

template <int H>
__global__ void edge_pass1(const int* __restrict__ ei, int E, int Etot,
                           const float* __restrict__ s, const float* __restrict__ d,
                           float* __restrict__ elog, unsigned* __restrict__ m)
{
  long e = (long)blockIdx.x * blockDim.x + threadIdx.x;
  if (e >= Etot) return;
  int src, dst; edge_sd(ei, E, e, src, dst);
#pragma unroll
  for (int hh = 0; hh < H; ++hh) {
    float t = s[(size_t)src * H + hh] + d[(size_t)dst * H + hh];
    t = t > 0.f ? t : 0.2f * t;                       // leaky_relu
    elog[(size_t)e * H + hh] = t;
    atomicMax(&m[(size_t)dst * H + hh], f2ord(t));    // segment_max
  }
}

template <int H>
__global__ void edge_pass2(const int* __restrict__ ei, int E, int Etot,
                           const unsigned* __restrict__ m, float* __restrict__ elog,
                           float* __restrict__ z)
{
  long e = (long)blockIdx.x * blockDim.x + threadIdx.x;
  if (e >= Etot) return;
  int src, dst; edge_sd(ei, E, e, src, dst);
  (void)src;
#pragma unroll
  for (int hh = 0; hh < H; ++hh) {
    float mf = ord2f(m[(size_t)dst * H + hh]);
    float w  = __expf(elog[(size_t)e * H + hh] - mf);
    elog[(size_t)e * H + hh] = w;
    atomicAdd(&z[(size_t)dst * H + hh], w);           // segment_sum of exp
  }
}

template <int H, int CS>
__global__ void edge_pass3(const int* __restrict__ ei, int E, int Etot,
                           const float* __restrict__ elog, const float* __restrict__ z,
                           const float* __restrict__ h, float* __restrict__ acc)
{
  long e = (long)blockIdx.x * blockDim.x + threadIdx.x;
  if (e >= Etot) return;
  int src, dst; edge_sd(ei, E, e, src, dst);
  float alpha[H];
#pragma unroll
  for (int hh = 0; hh < H; ++hh)
    alpha[hh] = elog[(size_t)e * H + hh] / z[(size_t)dst * H + hh];
  const float* hs = h + (size_t)src * 32;
#pragma unroll
  for (int j = 0; j < 32; ++j)
    atomicAdd(&acc[(size_t)dst * 32 + j], alpha[j >> CS] * hs[j]);  // weighted scatter
}

__global__ void finalize_kernel(const float* __restrict__ acc, const float* __restrict__ b,
                                float* __restrict__ out, int N, int do_elu)
{
  long i = (long)blockIdx.x * blockDim.x + threadIdx.x;
  if (i >= (long)N * 32) return;
  int j = (int)(i & 31);
  float v = acc[i] + b[j];
  if (do_elu) v = v > 0.f ? v : (__expf(v) - 1.f);
  out[i] = v;
}

__global__ void yf_kernel(const float* __restrict__ ys, const float* __restrict__ Wfin,
                          float* __restrict__ yf, int R)
{
  int r = blockIdx.x * blockDim.x + threadIdx.x;
  if (r >= R) return;
  float a = 0.f;
#pragma unroll
  for (int c = 0; c < 32; ++c) a += ys[(size_t)r * 32 + c] * Wfin[c];
  yf[r] = a;
}

// attention pooling + sigmoid; ys tile staged in LDS
__global__ __launch_bounds__(256) void pool_kernel(
    const float* __restrict__ x, const float* __restrict__ ys,
    const float* __restrict__ yf, const float* __restrict__ bfin,
    float* __restrict__ out, int nPerB, int F, int blocksPerB)
{
  __shared__ float ysh[64 * 32];
  __shared__ float yfh[64];
  int b  = blockIdx.x / blocksPerB;
  int n0 = (blockIdx.x % blocksPerB) * blockDim.x;
  for (int t = threadIdx.x; t < F * 32; t += blockDim.x)
    ysh[t] = ys[((size_t)b * F) * 32 + t];
  for (int t = threadIdx.x; t < F; t += blockDim.x)
    yfh[t] = yf[b * F + t];
  __syncthreads();
  int n = n0 + threadIdx.x;
  if (n >= nPerB) return;
  const float* xr = x + ((size_t)b * nPerB + n) * 32;
  float xv[32];
#pragma unroll
  for (int c = 0; c < 32; ++c) xv[c] = xr[c];
  float mx = -__builtin_inff();
  for (int f = 0; f < F; ++f) {
    float l = 0.f;
#pragma unroll
    for (int c = 0; c < 32; ++c) l += xv[c] * ysh[f * 32 + c];
    mx = fmaxf(mx, l);
  }
  float sum = 0.f, accv = 0.f;
  for (int f = 0; f < F; ++f) {
    float l = 0.f;
#pragma unroll
    for (int c = 0; c < 32; ++c) l += xv[c] * ysh[f * 32 + c];
    float w = __expf(l - mx);
    sum += w; accv += w * yfh[f];
  }
  float pv = accv / sum + bfin[0];
  out[(size_t)b * nPerB + n] = 1.f / (1.f + __expf(-pv));
}

// ======================= host orchestration =======================
extern "C" void kernel_launch(void* const* d_in, const int* in_sizes, int n_in,
                              void* d_out, int out_size, void* d_ws, size_t ws_size,
                              hipStream_t stream)
{
  const float* feats = (const float*)d_in[0];
  const int*   ei    = (const int*)d_in[1];
  const float* frag  = (const float*)d_in[3];
  const float* W1 = (const float*)d_in[4];   const float* b1 = (const float*)d_in[5];
  const float* W2 = (const float*)d_in[6];   const float* b2 = (const float*)d_in[7];
  const float* Wc = (const float*)d_in[8];   const float* bc = (const float*)d_in[9];
  const float* g1W  = (const float*)d_in[10];
  const float* g1as = (const float*)d_in[11];
  const float* g1ad = (const float*)d_in[12];
  const float* g1b  = (const float*)d_in[13];
  const float* g2W  = (const float*)d_in[14];
  const float* g2as = (const float*)d_in[15];
  const float* g2ad = (const float*)d_in[16];
  const float* g2b  = (const float*)d_in[17];
  const float* Wfin = (const float*)d_in[18];
  const float* bfin = (const float*)d_in[19];

  const int NF = 1304, M1 = 652, M2 = 326, M3 = 350, CH = 32, L = 3, Ff = 50;
  const int N    = in_sizes[0] / NF;             // 100000
  const int E    = in_sizes[1] / 2;              // 3200000
  const int Bb   = in_sizes[3] / (Ff * NF);      // 8
  const int R    = Bb * Ff;                      // 400 fragment rows
  const int Etot = E + N;

  // K-steps and tile counts; NT padded to multiple of NTW (pad tiles are zeros)
  const int K1 = 1280, KS1 = 40, NP1 = 672, NT1 = 44;   // 672 cols -> 42 tiles, pad 44
  const int KS2 = 21,  NT2 = 24;                        // K=672, 326 cols -> 21, pad 24
  const int KS3 = 11,  NT3 = 2;                         // K=352, 32 cols -> 2
  // ----- workspace carve-up -----
  char* w = (char*)d_ws; size_t off = 0;
  auto alloc = [&](size_t bytes) -> char* {
    char* p = w + off; off = (off + bytes + 255) & ~(size_t)255; return p;
  };
  float*    b1p  = (float*)alloc((size_t)NP1 * 4);
  bf16*     pW1  = (bf16*)alloc((size_t)NT1 * KS1 * 512 * 2);
  bf16*     pW2  = (bf16*)alloc((size_t)NT2 * KS2 * 512 * 2);
  bf16*     pWc  = (bf16*)alloc((size_t)NT3 * KS3 * 512 * 2);
  bf16*     pG1  = (bf16*)alloc((size_t)L * 1024 * 2);
  bf16*     pG2  = (bf16*)alloc((size_t)L * 1024 * 2);
  bf16*     esm1 = (bf16*)alloc((size_t)N * NP1 * 2);
  bf16*     x350 = (bf16*)alloc((size_t)N * 352 * 2);
  float*    xA   = (float*)alloc((size_t)N * 32 * 4);
  float*    x1   = (float*)alloc((size_t)N * 32 * 4);
  float*    hb   = (float*)alloc((size_t)N * 32 * 4);
  float*    sb   = (float*)alloc((size_t)N * 8 * 4);
  float*    db   = (float*)alloc((size_t)N * 8 * 4);
  unsigned* mb   = (unsigned*)alloc((size_t)N * 8 * 4);
  float*    zb   = (float*)alloc((size_t)N * 8 * 4);
  float*    accb = (float*)alloc((size_t)N * 32 * 4);
  float*    elog = (float*)alloc((size_t)Etot * 8 * 4);
  float*    ysb  = (float*)alloc((size_t)R * 32 * 4);
  float*    yfb  = (float*)alloc((size_t)R * 4);
  (void)ws_size; (void)n_in; (void)out_size;

  // ----- pack weights (once per launch; tiny) -----
  padbias_kernel<<<(NP1 + 255) / 256, 256, 0, stream>>>(b1, b1p, M1, NP1);
  { int tot = NT1 * KS1 * 512;
    pack_b_kernel<<<(tot + 255) / 256, 256, 0, stream>>>(W1, pW1, K1, M1, M1, KS1, NT1); }
  { int tot = NT2 * KS2 * 512;
    pack_b_kernel<<<(tot + 255) / 256, 256, 0, stream>>>(W2, pW2, M1, M2, M2, KS2, NT2); }
  { int tot = NT3 * KS3 * 512;
    pack_b_kernel<<<(tot + 255) / 256, 256, 0, stream>>>(Wc, pWc, M3, CH, CH, KS3, NT3); }
  for (int l = 0; l < L; ++l) {
    pack_b_kernel<<<4, 256, 0, stream>>>(g1W + (size_t)l * 1024, pG1 + (size_t)l * 1024,
                                         32, 32, 32, 1, 2);
    pack_b_kernel<<<4, 256, 0, stream>>>(g2W + (size_t)l * 1024, pG2 + (size_t)l * 1024,
                                         32, 32, 32, 1, 2);
  }

  dim3 blk(128);
  auto run_embed = [&](const float* src, int M, float* xout) {
    dim3 gA((M + 63) / 64, NT1 / 4);
    gemm_wmma<float, bf16, 4><<<gA, blk, 0, stream>>>(src + 24, NF, pW1, KS1,
                                                      b1p, esm1, NP1, M, NP1);
    build24_kernel<<<((M * 32) + 255) / 256, 256, 0, stream>>>(src, NF, x350, M);
    dim3 gB((M + 63) / 64, NT2 / 4);
    gemm_wmma<bf16, bf16, 4><<<gB, blk, 0, stream>>>(esm1, NP1, pW2, KS2,
                                                     b2, x350 + 24, 352, M, M2);
    dim3 gC((M + 63) / 64, NT3 / 2);
    gemm_wmma<bf16, float, 2><<<gC, blk, 0, stream>>>(x350, 352, pWc, KS3,
                                                      bc, xout, 32, M, CH);
  };

  auto run_conv = [&](const float* xin, const bf16* Wp, const float* as, const float* ad,
                      const float* bias, int H, int do_elu, float* xout) {
    dim3 gg((N + 63) / 64, 1);
    gemm_wmma<float, float, 2><<<gg, blk, 0, stream>>>(xin, 32, Wp, 1,
                                                       nullptr, hb, 32, N, 32);
    sd_init_kernel<<<(N + 255) / 256, 256, 0, stream>>>(hb, as, ad, sb, db, mb, zb,
                                                        accb, N, H);
    int eb = (Etot + 255) / 256;
    if (H == 8) {
      edge_pass1<8><<<eb, 256, 0, stream>>>(ei, E, Etot, sb, db, elog, mb);
      edge_pass2<8><<<eb, 256, 0, stream>>>(ei, E, Etot, mb, elog, zb);
      edge_pass3<8, 2><<<eb, 256, 0, stream>>>(ei, E, Etot, elog, zb, hb, accb);
    } else {
      edge_pass1<1><<<eb, 256, 0, stream>>>(ei, E, Etot, sb, db, elog, mb);
      edge_pass2<1><<<eb, 256, 0, stream>>>(ei, E, Etot, mb, elog, zb);
      edge_pass3<1, 5><<<eb, 256, 0, stream>>>(ei, E, Etot, elog, zb, hb, accb);
    }
    finalize_kernel<<<(int)(((long)N * 32 + 255) / 256), 256, 0, stream>>>(
        accb, bias, xout, N, do_elu);
  };

  // 1) embed nodes
  run_embed(feats, N, xA);

  // 2) 3x (GATConv 8 heads + ELU, GATConv 1 head)
  for (int l = 0; l < L; ++l) {
    run_conv(xA, pG1 + (size_t)l * 1024, g1as + (size_t)l * 32, g1ad + (size_t)l * 32,
             g1b + (size_t)l * 32, 8, 1, x1);
    run_conv(x1, pG2 + (size_t)l * 1024, g2as + (size_t)l * 32, g2ad + (size_t)l * 32,
             g2b + (size_t)l * 32, 1, 0, xA);
  }

  // 3) embed fragments (reuses esm1/x350 scratch)
  run_embed(frag, R, ysb);
  yf_kernel<<<(R + 255) / 256, 256, 0, stream>>>(ysb, Wfin, yfb, R);

  // 4) attention pooling + sigmoid
  int nPerB = N / Bb;
  int bpb = (nPerB + 255) / 256;
  pool_kernel<<<Bb * bpb, 256, 0, stream>>>(xA, ysb, yfb, bfin, (float*)d_out,
                                            nPerB, Ff, bpb);
}
